// MptAttention_88313117540434
// MI455X (gfx1250) — matrix operand order repaired and checked by reading, not compiled, not run on hardware
//
#include <hip/hip_runtime.h>

typedef __bf16 bf16;
typedef __attribute__((ext_vector_type(16))) __bf16 v16bf;
typedef __attribute__((ext_vector_type(8)))  float  v8f;
typedef __attribute__((ext_vector_type(4)))  __bf16 v4bf;
typedef int v4i_vs __attribute__((vector_size(16)));  // matches builtin param type

#if defined(__has_builtin)
#  if __has_builtin(__builtin_amdgcn_global_load_async_to_lds_b128)
#    define HAVE_ASYNC_LDS 1
#  endif
#endif
#ifndef HAVE_ASYNC_LDS
#  define HAVE_ASYNC_LDS 0
#endif

union Frag16 { uint4 u[2]; v16bf v; };

__device__ __forceinline__ v8f v8f_zero() {
  v8f z;
#pragma unroll
  for (int i = 0; i < 8; ++i) z[i] = 0.0f;
  return z;
}

__device__ __forceinline__ v8f wmma_bf16(v16bf a, v16bf b, v8f c) {
  // emits v_wmma_f32_16x16x32_bf16
  return __builtin_amdgcn_wmma_f32_16x16x32_bf16(false, a, false, b, (short)0, c,
                                                 false, false);
}

// 8 bf16 (16B) global -> LDS copy; async (ASYNCcnt) on CDNA5 when available.
__device__ __forceinline__ void copy_b128_to_lds(const bf16* g, bf16* l) {
#if HAVE_ASYNC_LDS
  __builtin_amdgcn_global_load_async_to_lds_b128(
      (__attribute__((address_space(1))) v4i_vs*)(g),
      (__attribute__((address_space(3))) v4i_vs*)(l), 0, 0);
#else
  *(uint4*)l = *(const uint4*)g;
#endif
}

__device__ __forceinline__ void wait_async_copies() {
#if HAVE_ASYNC_LDS
#  if defined(__has_builtin)
#    if __has_builtin(__builtin_amdgcn_s_wait_asynccnt)
  __builtin_amdgcn_s_wait_asynccnt(0);
#    else
  asm volatile("s_wait_asynccnt 0x0" ::: "memory");
#    endif
#  else
  asm volatile("s_wait_asynccnt 0x0" ::: "memory");
#  endif
#endif
}

// A fragment: 16x32 (MxK) from row-major [*, ld] storage.
// lane (m = l&15, h = l>>4): elems 0..7 = (m, 8h+0..7), elems 8..15 = (m, 16+8h+0..7)
__device__ __forceinline__ v16bf load_frag_a(const bf16* base, int ld) {
  const int lane = threadIdx.x & 31;
  const int m = lane & 15, h = lane >> 4;
  const bf16* p = base + m * ld + 8 * h;
  Frag16 f;
  f.u[0] = *(const uint4*)(p);
  f.u[1] = *(const uint4*)(p + 16);
  return f.v;
}

// B fragment: consumed as B(n,k), source row-major [N, ld] (rows = n, contiguous k).
// lane (n = l&15, h = l>>4): elems 0..7 = (n, 8h+0..7), elems 8..15 = (n, 16+8h+0..7)
__device__ __forceinline__ v16bf load_frag_b(const bf16* base, int ld) {
  const int lane = threadIdx.x & 31;
  const int n = lane & 15, h = lane >> 4;
  const bf16* p = base + n * ld + 8 * h;
  Frag16 f;
  f.u[0] = *(const uint4*)(p);
  f.u[1] = *(const uint4*)(p + 16);
  return f.v;
}

// ---------------- f32 -> bf16 convert (vectorized) ----------------
__global__ void cvt_f32_to_bf16(const float* __restrict__ in,
                                bf16* __restrict__ out, long n4) {
  long i = (long)blockIdx.x * blockDim.x + threadIdx.x;
  const long stride = (long)gridDim.x * blockDim.x;
  for (; i < n4; i += stride) {
    float4 f = ((const float4*)in)[i];
    v4bf o;
    o[0] = (bf16)f.x; o[1] = (bf16)f.y; o[2] = (bf16)f.z; o[3] = (bf16)f.w;
    ((v4bf*)out)[i] = o;
  }
}

// ---------------- Tiled WMMA GEMM:  C[M,N] = A[M,K] * B[N,K]^T + bias ----------------
// Block 256 thr (8 waves), tile 128(M) x 128(N) x 32(K). Wave = 32x64 = 2x4 WMMA tiles.
// Double-buffered LDS staging via async global->LDS copies (overlap with WMMA).
template <bool OUT_BF16>
__global__ __launch_bounds__(256) void gemm_bf16_kernel(
    const bf16* __restrict__ A, const bf16* __restrict__ B,
    const float* __restrict__ bias, float* __restrict__ Cf,
    bf16* __restrict__ Cb, int M, int N, int K) {
  __shared__ bf16 As[2][128][40];
  __shared__ bf16 Bs[2][128][40];
  const int tid = threadIdx.x;
  const int wid = tid >> 5;
  const int wm = wid >> 1;  // 0..3
  const int wn = wid & 1;   // 0..1
  const int bm = blockIdx.y * 128;
  const int bn = blockIdx.x * 128;

  v8f acc[2][4];
#pragma unroll
  for (int i = 0; i < 2; ++i)
#pragma unroll
    for (int j = 0; j < 4; ++j) acc[i][j] = v8f_zero();

  auto issue_tile = [&](int buf, int k0) {
#pragma unroll
    for (int i = 0; i < 2; ++i) {
      int g = tid + i * 256;    // 0..511 groups of 8 bf16
      int r = g >> 2;           // 0..127
      int cg = (g & 3) * 8;     // 0,8,16,24
      copy_b128_to_lds(&A[(size_t)(bm + r) * K + k0 + cg], &As[buf][r][cg]);
      copy_b128_to_lds(&B[(size_t)(bn + r) * K + k0 + cg], &Bs[buf][r][cg]);
    }
  };

  issue_tile(0, 0);
  int cur = 0;
  for (int k0 = 0; k0 < K; k0 += 32) {
    wait_async_copies();
    __syncthreads();                       // tile[cur] visible to all waves
    if (k0 + 32 < K) issue_tile(cur ^ 1, k0 + 32);  // prefetch next tile

    v16bf af[2], bfr[4];
#pragma unroll
    for (int i = 0; i < 2; ++i)
      af[i] = load_frag_a(&As[cur][wm * 32 + i * 16][0], 40);
#pragma unroll
    for (int j = 0; j < 4; ++j)
      bfr[j] = load_frag_b(&Bs[cur][wn * 64 + j * 16][0], 40);
#pragma unroll
    for (int i = 0; i < 2; ++i)
#pragma unroll
      for (int j = 0; j < 4; ++j) acc[i][j] = wmma_bf16(af[i], bfr[j], acc[i][j]);
    cur ^= 1;
  }

  const int lane = tid & 31;
  const int nl = lane & 15, h = lane >> 4;
#pragma unroll
  for (int j = 0; j < 4; ++j) {
    const int col = bn + wn * 64 + j * 16 + nl;
    const float bv = bias[col];
#pragma unroll
    for (int i = 0; i < 2; ++i) {
      const int row0 = bm + wm * 32 + i * 16 + 8 * h;
#pragma unroll
      for (int r = 0; r < 8; ++r) {
        const float v = acc[i][j][r] + bv;
        const size_t idx = (size_t)(row0 + r) * N + col;
        if (OUT_BF16) Cb[idx] = (bf16)v; else Cf[idx] = v;
      }
    }
  }
}

// ---------------- Flash attention (causal + ALiBi) ----------------
// grid (S/64, H); block 128 (4 waves). Wave owns 16 query rows, hd = 128.
__global__ __launch_bounds__(128) void flash_attn_kernel(
    const bf16* __restrict__ qkv,  // [S, 3D] bf16
    bf16* __restrict__ outb,       // [S, D]  bf16
    int S, int D) {
  const int D3 = 3 * D;
  __shared__ bf16 Ks[32][136];      // keys x d   (row-major, as B(n=key,k=d))
  __shared__ bf16 Vt[128][48];      // d x keys   (transposed, as B(n=d,k=key))
  __shared__ bf16 Ps[4][16][40];    // per-wave P staging (q x key)

  const int h = blockIdx.y;
  const int qb = blockIdx.x;
  const int tid = threadIdx.x;
  const int wid = tid >> 5;
  const int lane = tid & 31;
  const int nl = lane & 15, hh = lane >> 4;
  const int qw0 = qb * 64 + wid * 16;

  const float scale = 0.08838834764831845f;          // 1/sqrt(128)
  const float slope = exp2f(-0.25f * (float)(h + 1)); // ALiBi, H=32 pow2

  // Q fragments (persist in registers): 16 rows x 128 d = 4 x (16x32)
  v16bf qf[4];
  const bf16* qbase = qkv + (size_t)qw0 * D3 + h * 128;
#pragma unroll
  for (int c = 0; c < 4; ++c) qf[c] = load_frag_a(qbase + c * 32, D3);

  v8f o[8];
#pragma unroll
  for (int t = 0; t < 8; ++t) o[t] = v8f_zero();
  float mrow[8], lrow[8];
#pragma unroll
  for (int j = 0; j < 8; ++j) { mrow[j] = -1e30f; lrow[j] = 0.0f; }

  const int nkb = 2 * qb + 2;  // key blocks of 32 covering causal span
  for (int kb = 0; kb < nkb; ++kb) {
    const int k0 = kb * 32;
    // stage K (async, row-major) and V (sync, transposed) tiles into LDS
#pragma unroll
    for (int i = 0; i < 4; ++i) {
      int g = tid + i * 128;   // 0..511 groups of 8 bf16
      int r = g >> 4;          // key 0..31
      int cg = (g & 15) * 8;   // d offset 0..120
      copy_b128_to_lds(&qkv[(size_t)(k0 + r) * D3 + D + h * 128 + cg], &Ks[r][cg]);
      union { uint4 u; bf16 e[8]; } vv;
      vv.u = *(const uint4*)&qkv[(size_t)(k0 + r) * D3 + 2 * D + h * 128 + cg];
#pragma unroll
      for (int e = 0; e < 8; ++e) Vt[cg + e][r] = vv.e[e];
    }
    wait_async_copies();
    __syncthreads();

    // scores: 16 q x 32 keys, K-dim = hd = 128 (4 chained WMMAs per key tile)
    v8f s0 = v8f_zero(), s1 = v8f_zero();
#pragma unroll
    for (int c = 0; c < 4; ++c) {
      v16bf b0 = load_frag_b(&Ks[0][c * 32], 136);
      v16bf b1 = load_frag_b(&Ks[16][c * 32], 136);
      s0 = wmma_bf16(qf[c], b0, s0);
      s1 = wmma_bf16(qf[c], b1, s1);
    }

    // online softmax: row m = 8*hh + j lives in this 16-lane half, vgpr j
    float fct[8];
#pragma unroll
    for (int j = 0; j < 8; ++j) {
      const int ck0 = k0 + nl, ck1 = k0 + 16 + nl;
      const int rowq = qw0 + 8 * hh + j;
      float v0 = s0[j] * scale + slope * (float)(ck0 - (S - 1));
      float v1 = s1[j] * scale + slope * (float)(ck1 - (S - 1));
      if (ck0 > rowq) v0 = -1e30f;
      if (ck1 > rowq) v1 = -1e30f;
      float t = fmaxf(v0, v1);
#pragma unroll
      for (int off = 1; off < 16; off <<= 1) t = fmaxf(t, __shfl_xor(t, off, 32));
      const float mnew = fmaxf(mrow[j], t);
      const float f = __expf(mrow[j] - mnew);
      v0 = __expf(v0 - mnew);
      v1 = __expf(v1 - mnew);
      float rs = v0 + v1;
#pragma unroll
      for (int off = 1; off < 16; off <<= 1) rs += __shfl_xor(rs, off, 32);
      lrow[j] = lrow[j] * f + rs;
      mrow[j] = mnew;
      fct[j] = f;
      s0[j] = v0;
      s1[j] = v1;
    }
#pragma unroll
    for (int t = 0; t < 8; ++t)
#pragma unroll
      for (int j = 0; j < 8; ++j) o[t][j] *= fct[j];

    // C-layout P -> LDS -> A-layout fragment (wave-private; DS is in-order per wave)
#pragma unroll
    for (int j = 0; j < 8; ++j) {
      Ps[wid][8 * hh + j][nl] = (bf16)s0[j];
      Ps[wid][8 * hh + j][16 + nl] = (bf16)s1[j];
    }
    asm volatile("s_wait_dscnt 0x0" ::: "memory");
    v16bf pf = load_frag_a(&Ps[wid][0][0], 40);
#pragma unroll
    for (int t = 0; t < 8; ++t) {
      v16bf vf = load_frag_b(&Vt[t * 16][0], 48);
      o[t] = wmma_bf16(pf, vf, o[t]);
    }
    __syncthreads();
  }

  // normalize + store bf16 [S, D]
#pragma unroll
  for (int t = 0; t < 8; ++t) {
    const int col = h * 128 + t * 16 + nl;
#pragma unroll
    for (int j = 0; j < 8; ++j) {
      const int row = qw0 + 8 * hh + j;
      outb[(size_t)row * D + col] = (bf16)(o[t][j] / lrow[j]);
    }
  }
}

// ---------------- launcher ----------------
extern "C" void kernel_launch(void* const* d_in, const int* in_sizes, int n_in,
                              void* d_out, int out_size, void* d_ws, size_t ws_size,
                              hipStream_t stream) {
  (void)in_sizes; (void)n_in; (void)out_size; (void)ws_size;
  const float* x     = (const float*)d_in[0];  // [1,2048,4096]
  const float* w_qkv = (const float*)d_in[1];  // [12288,4096]
  const float* b_qkv = (const float*)d_in[2];  // [12288]
  const float* w_out = (const float*)d_in[3];  // [4096,4096]
  const float* b_out = (const float*)d_in[4];  // [4096]
  float* out = (float*)d_out;                  // [1,2048,4096] f32

  const int S = 2048, D = 4096, D3 = 12288;

  bf16* xb   = (bf16*)d_ws;                      // [S,D]
  bf16* wqb  = xb + (size_t)S * D;               // [3D,D]
  bf16* wob  = wqb + (size_t)D3 * D;             // [D,D]
  bf16* qkvb = wob + (size_t)D * D;              // [S,3D]
  bf16* aob  = qkvb + (size_t)S * D3;            // [S,D]

  // 1) downconvert fp32 -> bf16
  cvt_f32_to_bf16<<<2048, 256, 0, stream>>>(x, xb, (long)S * D / 4);
  cvt_f32_to_bf16<<<2048, 256, 0, stream>>>(w_qkv, wqb, (long)D3 * D / 4);
  cvt_f32_to_bf16<<<2048, 256, 0, stream>>>(w_out, wob, (long)D * D / 4);

  // 2) QKV projection: [S,D] x [3D,D]^T + b_qkv -> bf16 [S,3D]
  gemm_bf16_kernel<true><<<dim3(D3 / 128, S / 128), 256, 0, stream>>>(
      xb, wqb, b_qkv, nullptr, qkvb, S, D3, D);

  // 3) causal + ALiBi flash attention -> bf16 [S,D]
  flash_attn_kernel<<<dim3(S / 64, 32), 128, 0, stream>>>(qkvb, aob, S, D);

  // 4) output projection: [S,D] x [D,D]^T + b_out -> f32 d_out
  gemm_bf16_kernel<false><<<dim3(D / 128, S / 128), 256, 0, stream>>>(
      aob, wob, b_out, out, nullptr, S, D, D);
}
